// GeoGaussNO_32719060861296
// MI455X (gfx1250) — matrix operand order repaired and checked by reading, not compiled, not run on hardware
//
#include <hip/hip_runtime.h>

// Problem constants (from the reference)
#define BB   2
#define NNODE 16384
#define EEDGE 131072
#define CCH  128
#define KMODE 64
#define INF  3
#define FCF  128
#define LLAY 4

typedef __attribute__((ext_vector_type(16))) __bf16 v16bf;
typedef __attribute__((ext_vector_type(8)))  __bf16 v8bf;
typedef __attribute__((ext_vector_type(8)))  float  v8f;

__device__ __forceinline__ float gelu_f(float x) {
  return 0.5f * x * (1.0f + erff(x * 0.70710678118654752440f));
}

// ---------------------------------------------------------------------------
// Generic tiled bf16 GEMM with f32 accumulate via v_wmma_f32_16x16x32_bf16.
// D[b] (M x N, ld=N, f32) = op( alpha * A[b](M x Kc, row-major bf16) *
//                               B[b](Kc x N, row-major bf16) )
// mode: 0 = store (opt bias/act), 1 = accumulate-read (opt bias/act),
//       2 = atomicAdd (split-K; no bias/act)
// Requires: M,N multiples of 64; Kc/ksplit multiple of 32; A/B buffers
// 16B-aligned with even leading dims (true for all uses here).
//
// LDS layouts are fragment-ready:
//   As row-major   As[r*32 + k]  -> A frag = two contiguous 16B runs per lane
//   Bs col-major   Bs[n*32 + k]  -> B frag = one contiguous 32B run per lane
// so every fragment fetch is ds_load_b128; staging issues all four b128
// global loads before any LDS store so they pipeline under one loadcnt wait.
// ---------------------------------------------------------------------------
#define GBM 64
#define GBN 64
#define GBK 32

__global__ __launch_bounds__(128)
void gemm_bf16(const __bf16* __restrict__ A, const __bf16* __restrict__ Bm,
               float* __restrict__ D, const float* __restrict__ bias,
               int M, int N, int Kc,
               long strideA, long strideB, long strideD,
               float alpha, int mode, int act, int ksplit)
{
  __shared__ __align__(32) __bf16 As[GBM * GBK];
  __shared__ __align__(32) __bf16 Bs[GBN * GBK];

  const int tid  = threadIdx.x;
  const int wave = tid >> 5;
  const int lane = tid & 31;
  const int b    = blockIdx.z / ksplit;
  const int kid  = blockIdx.z % ksplit;
  const int row0 = blockIdx.y * GBM;
  const int col0 = blockIdx.x * GBN;
  const int klen = Kc / ksplit;
  const int kbeg = kid * klen;

  const __bf16* Ab = A  + (long)b * strideA;
  const __bf16* Bb = Bm + (long)b * strideB;
  float*        Db = D  + (long)b * strideD;

  v8f acc[4];
#pragma unroll
  for (int t = 0; t < 4; ++t)
#pragma unroll
    for (int v = 0; v < 8; ++v) acc[t][v] = 0.0f;

  const int lm = lane & 15;   // row/col within a 16-wide tile
  const int lh = lane >> 4;   // lane-half selector

  // Staging coordinates (compile-time trip counts, no guards).
  const int ar0 = tid >> 2;          // A row, first half (0..31)
  const int aq  = tid & 3;           // A 16B-quad within row
  const int bkp = tid >> 3;          // B k-pair (0..15)
  const int bn8 = (tid & 7) << 3;    // B col base, 8 wide

  for (int kk = kbeg; kk < kbeg + klen; kk += GBK) {
    // Issue all four independent b128 global loads first...
    uint4 ua0 = *(const uint4*)(Ab + (long)(row0 + ar0     ) * Kc + kk + (aq << 3));
    uint4 ua1 = *(const uint4*)(Ab + (long)(row0 + ar0 + 32) * Kc + kk + (aq << 3));
    const __bf16* bsrc = Bb + (long)(kk + 2 * bkp) * N + col0 + bn8;
    uint4 ub0 = *(const uint4*)(bsrc);
    uint4 ub1 = *(const uint4*)(bsrc + N);
    if (kk + GBK < kbeg + klen)  // hint next A panel -> global_prefetch_b8
      __builtin_prefetch(Ab + (long)(row0 + ar0) * Kc + kk + GBK, 0, 0);

    // ...then store to LDS. A panel: row-major, two b128 stores.
    *(uint4*)(&As[( ar0       << 5) + (aq << 3)]) = ua0;
    *(uint4*)(&As[((ar0 + 32) << 5) + (aq << 3)]) = ua1;
    // B panel: transpose k-pairs into col-major LDS ({B[k][n],B[k+1][n]}).
    {
      const unsigned* lo = (const unsigned*)&ub0;
      const unsigned* hi = (const unsigned*)&ub1;
#pragma unroll
      for (int j = 0; j < 4; ++j) {
        unsigned w0 = (lo[j] & 0xFFFFu) | (hi[j] << 16);
        unsigned w1 = (lo[j] >> 16) | (hi[j] & 0xFFFF0000u);
        *(unsigned*)(&Bs[(bn8 + 2 * j    ) * GBK + (bkp << 1)]) = w0;
        *(unsigned*)(&Bs[(bn8 + 2 * j + 1) * GBK + (bkp << 1)]) = w1;
      }
    }
    __syncthreads();

    // A fragment: lane lm = M row in this wave's 16 rows; element j maps to
    // k = lh*16 + j: two contiguous 16B runs.
    const __bf16* arow = &As[((wave << 4) + lm) * GBK + (lh << 4)];
    v8bf a0 = *(const v8bf*)(arow);
    v8bf a1 = *(const v8bf*)(arow + 8);
    v16bf afrag = __builtin_shufflevector(a0, a1,
        0, 1, 2, 3, 4, 5, 6, 7, 8, 9, 10, 11, 12, 13, 14, 15);

#pragma unroll
    for (int t = 0; t < 4; ++t) {
      // B fragment: element j maps to k = lh*16 + j at col n: one 32B run.
      v16bf bfrag = *(const v16bf*)(&Bs[((t << 4) + lm) * GBK + (lh << 4)]);
      acc[t] = __builtin_amdgcn_wmma_f32_16x16x32_bf16(
          false, afrag, false, bfrag, (short)0, acc[t], false, false);
    }
    __syncthreads();
  }

  // Writeback: C/D layout -> row = v + 8*lh, col = lm (per 16x16 tile).
#pragma unroll
  for (int t = 0; t < 4; ++t) {
#pragma unroll
    for (int v = 0; v < 8; ++v) {
      int row = row0 + (wave << 4) + v + (lh << 3);
      int col = col0 + (t << 4) + lm;
      long off = (long)row * N + col;
      float val = alpha * acc[t][v];
      if (mode == 2) {
        atomicAdd(Db + off, val);
      } else {
        if (bias) val += bias[row];
        if (mode == 1) val += Db[off];
        if (act) val = gelu_f(val);
        Db[off] = val;
      }
    }
  }
}

// ---------------------------------------------------------------------------
// Small helper kernels
// ---------------------------------------------------------------------------
__global__ void k_zero(float* pdata, long n) {
  long i = (long)blockIdx.x * blockDim.x + threadIdx.x;
  long stride = (long)gridDim.x * blockDim.x;
  for (; i < n; i += stride) pdata[i] = 0.0f;
}

__global__ void k_cvt(const float* __restrict__ in, __bf16* __restrict__ out, long n) {
  long i = (long)blockIdx.x * blockDim.x + threadIdx.x;
  if (i < n) out[i] = (__bf16)in[i];
}

// fc1_w (C,FC) -> bf16 transposed (FC,C)
__global__ void k_cvt_fc1T(const float* __restrict__ in, __bf16* __restrict__ out) {
  int i = blockIdx.x * blockDim.x + threadIdx.x;
  if (i < CCH * FCF) {
    int c = i / FCF, f = i % FCF;
    out[f * CCH + c] = (__bf16)in[i];
  }
}

// h (B,C,N) f32 -> hb bf16; optionally hwb = bf16(h * node_weights[b,n])
__global__ void k_cvt_h(const float* __restrict__ h, const float* __restrict__ wt,
                        __bf16* __restrict__ hb, __bf16* __restrict__ hwb, long total) {
  long i = (long)blockIdx.x * blockDim.x + threadIdx.x;
  if (i >= total) return;
  float v = h[i];
  hb[i] = (__bf16)v;
  if (hwb) {
    long n = i % NNODE;
    long b = i / ((long)CCH * NNODE);
    hwb[i] = (__bf16)(v * wt[b * NNODE + n]);
  }
}

// Fourier bases (cos/sin of nodes . modes) * mask, in both (N,K) and (K,N)
__global__ __launch_bounds__(256)
void k_bases(const float* __restrict__ nodes, const float* __restrict__ mask,
             const float* __restrict__ modes,
             __bf16* bc_nk, __bf16* bs_nk, __bf16* bc_kn, __bf16* bs_kn) {
  __shared__ float md[2 * KMODE];
  if (threadIdx.x < 2 * KMODE) md[threadIdx.x] = modes[threadIdx.x];
  __syncthreads();
  int n = blockIdx.x * 256 + threadIdx.x;
  int b = blockIdx.y;
  float nx = nodes[((long)b * NNODE + n) * 2];
  float ny = nodes[((long)b * NNODE + n) * 2 + 1];
  float mk = mask[(long)b * NNODE + n];
  for (int k = 0; k < KMODE; ++k) {
    float t = nx * md[2 * k] + ny * md[2 * k + 1];
    float c = cosf(t) * mk, s = sinf(t) * mk;
    bc_nk[((long)b * NNODE + n) * KMODE + k] = (__bf16)c;
    bs_nk[((long)b * NNODE + n) * KMODE + k] = (__bf16)s;
    bc_kn[((long)b * KMODE + k) * NNODE + n] = (__bf16)c;
    bs_kn[((long)b * KMODE + k) * NNODE + n] = (__bf16)s;
  }
}

// h = (x @ fc0_w + fc0_b) transposed to (B,C,N)
__global__ void k_fc0(const float* __restrict__ x, const float* __restrict__ w,
                      const float* __restrict__ bias, float* __restrict__ h) {
  long i = (long)blockIdx.x * blockDim.x + threadIdx.x;
  long total = (long)BB * CCH * NNODE;
  if (i >= total) return;
  long b = i / ((long)CCH * NNODE);
  long r = i % ((long)CCH * NNODE);
  int c = (int)(r / NNODE), n = (int)(r % NNODE);
  float acc = bias[c];
  for (int k = 0; k < INF; ++k)
    acc += x[((long)b * NNODE + n) * INF + k] * w[k * CCH + c];
  h[i] = acc;
}

// x0[b,c] = sum_n h[b,c,n]*wt[b,n]*mask[b,n]
__global__ void k_x0(const float* __restrict__ h, const float* __restrict__ wt,
                     const float* __restrict__ mask, float* __restrict__ x0) {
  __shared__ float red[256];
  int c = blockIdx.x, b = blockIdx.y;
  const float* hr = h + ((long)b * CCH + c) * NNODE;
  const float* wr = wt + (long)b * NNODE;
  const float* mr = mask + (long)b * NNODE;
  float acc = 0.0f;
  for (int n = threadIdx.x; n < NNODE; n += 256) acc += hr[n] * wr[n] * mr[n];
  red[threadIdx.x] = acc;
  __syncthreads();
  for (int s = 128; s > 0; s >>= 1) {
    if (threadIdx.x < s) red[threadIdx.x] += red[threadIdx.x + s];
    __syncthreads();
  }
  if (threadIdx.x == 0) x0[(long)b * CCH + c] = red[0];
}

// fc_h/fs_h complex spectral weight mix (per (b,k)); outputs bf16 (C,K)
__global__ __launch_bounds__(128)
void k_specmix(const float* __restrict__ xc, const float* __restrict__ xs,
               const float* __restrict__ wc, const float* __restrict__ ws,
               __bf16* __restrict__ fch, __bf16* __restrict__ fsh) {
  __shared__ float sc[CCH], ss[CCH];
  int k = blockIdx.x, b = blockIdx.y, o = threadIdx.x;
  sc[o] = xc[((long)b * CCH + o) * KMODE + k];
  ss[o] = xs[((long)b * CCH + o) * KMODE + k];
  __syncthreads();
  float fc = 0.0f, fs = 0.0f;
  for (int i = 0; i < CCH; ++i) {
    float c = wc[((long)i * CCH + o) * KMODE + k];
    float s = ws[((long)i * CCH + o) * KMODE + k];
    fc += sc[i] * c - ss[i] * s;
    fs += ss[i] * c + sc[i] * s;
  }
  fch[((long)b * CCH + o) * KMODE + k] = (__bf16)fc;
  fsh[((long)b * CCH + o) * KMODE + k] = (__bf16)fs;
}

// f0_h[b,o] = sum_i x0[b,i]*sp_w0[i,o]
__global__ void k_f0(const float* __restrict__ x0, const float* __restrict__ w0,
                     float* __restrict__ f0) {
  int b = blockIdx.x, o = threadIdx.x;
  float acc = 0.0f;
  for (int i = 0; i < CCH; ++i) acc += x0[(long)b * CCH + i] * w0[(long)i * CCH + o];
  f0[(long)b * CCH + o] = acc;
}

// (B,C,N) -> (B,N,C) via LDS 32x32 tiles
__global__ void k_transpose(const float* __restrict__ in, float* __restrict__ out) {
  __shared__ float tile[32][33];
  int b = blockIdx.z;
  int n0 = blockIdx.x * 32, c0 = blockIdx.y * 32;
  const float* inb = in + (long)b * CCH * NNODE;
  float* outb = out + (long)b * CCH * NNODE;
  for (int i = threadIdx.y; i < 32; i += 8)
    tile[i][threadIdx.x] = inb[(long)(c0 + i) * NNODE + n0 + threadIdx.x];
  __syncthreads();
  for (int i = threadIdx.y; i < 32; i += 8)
    outb[(long)(n0 + i) * CCH + c0 + threadIdx.x] = tile[threadIdx.x][i];
}

// Gaussian graph branch: gather f[src], weight by kernel, scatter-add to x3[tgt]
__global__ __launch_bounds__(128)
void k_edges(const int* __restrict__ edges, const float* __restrict__ nodes,
             const float* __restrict__ ft, const float* __restrict__ isw,
             const float* __restrict__ isb, const float* __restrict__ gc,
             float* __restrict__ x3) {
  int e = blockIdx.x, b = blockIdx.y;
  const int* ep = edges + ((long)b * EEDGE + e) * 2;
  int src = ep[0], tgt = ep[1];
  const float* ns = nodes + ((long)b * NNODE + src) * 2;
  const float* nt = nodes + ((long)b * NNODE + tgt) * 2;
  float e0 = ns[0] - nt[0], e1 = ns[1] - nt[1];
  float s0 = e0 * isw[0] + e1 * isw[1] + isb[0];
  float s1 = e0 * isw[2] + e1 * isw[3] + isb[1];
  float ker = gc[0] * expf(e0 * s0 + e1 * s1);
  int c = threadIdx.x;
  float msg = ft[((long)b * NNODE + src) * CCH + c] * ker;
  atomicAdd(&x3[((long)b * NNODE + tgt) * CCH + c], msg);
}

// hnew = (x1+x2) [+ graph]: h = gelu(d1 + f0*mask) + gelu(x3^T) or plain sum
__global__ void k_combine(const float* __restrict__ d1, const float* __restrict__ f0,
                          const float* __restrict__ mask, const float* __restrict__ x3,
                          float* __restrict__ h, int dograph) {
  long i = (long)blockIdx.x * blockDim.x + threadIdx.x;
  long total = (long)BB * CCH * NNODE;
  if (i >= total) return;
  long b = i / ((long)CCH * NNODE);
  long r = i % ((long)CCH * NNODE);
  int c = (int)(r / NNODE), n = (int)(r % NNODE);
  float val = d1[i] + f0[b * CCH + c] * mask[b * NNODE + n];
  if (dograph)
    val = gelu_f(val) + gelu_f(x3[((long)b * NNODE + n) * CCH + c]);
  h[i] = val;
}

// out[b,n] = sum_f h1[b,f,n]*fc2_w[f] + fc2_b
__global__ void k_out(const float* __restrict__ h1, const float* __restrict__ w2,
                      const float* __restrict__ b2, float* __restrict__ out) {
  int n = blockIdx.x * 256 + threadIdx.x;
  int b = blockIdx.y;
  float acc = b2[0];
  for (int f = 0; f < FCF; ++f) acc += h1[((long)b * FCF + f) * NNODE + n] * w2[f];
  out[(long)b * NNODE + n] = acc;
}

// ---------------------------------------------------------------------------
extern "C" void kernel_launch(void* const* d_in, const int* in_sizes, int n_in,
                              void* d_out, int out_size, void* d_ws, size_t ws_size,
                              hipStream_t stream) {
  (void)in_sizes; (void)n_in; (void)out_size; (void)ws_size;
  const float* x        = (const float*)d_in[0];
  const float* mask     = (const float*)d_in[1];
  const float* nodes    = (const float*)d_in[2];
  const float* nw       = (const float*)d_in[3];
  const int*   edges    = (const int*)d_in[4];
  const float* modes    = (const float*)d_in[5];
  const float* fc0_w    = (const float*)d_in[6];
  const float* fc0_b    = (const float*)d_in[7];
  const float* sp_wc    = (const float*)d_in[8];
  const float* sp_ws    = (const float*)d_in[9];
  const float* sp_w0    = (const float*)d_in[10];
  const float* w_w      = (const float*)d_in[11];
  const float* w_b      = (const float*)d_in[12];
  const float* g_w1     = (const float*)d_in[13];
  const float* g_b1     = (const float*)d_in[14];
  const float* g_w2     = (const float*)d_in[15];
  const float* g_b2     = (const float*)d_in[16];
  const float* isw      = (const float*)d_in[17];
  const float* isb      = (const float*)d_in[18];
  const float* gC       = (const float*)d_in[19];
  const float* fc1_w    = (const float*)d_in[20];
  const float* fc1_b    = (const float*)d_in[21];
  const float* fc2_w    = (const float*)d_in[22];
  const float* fc2_b    = (const float*)d_in[23];
  float* out = (float*)d_out;

  const long BCN = (long)BB * CCH * NNODE;       // 4,194,304
  const long BNK = (long)BB * NNODE * KMODE;     // 2,097,152
  const long BCK = (long)BB * CCH * KMODE;       // 16,384

  // Carve workspace (~145 MB total).
  char* p = (char*)d_ws;
  auto alloc = [&](size_t bytes) -> void* {
    void* r = (void*)p;
    p += (bytes + 255) & ~(size_t)255;
    return r;
  };
  float*  H    = (float*)alloc(BCN * 4);
  float*  D1   = (float*)alloc(BCN * 4);
  float*  MBUF = (float*)alloc(BCN * 4);   // also reused as H1 in final stage
  float*  FBUF = (float*)alloc(BCN * 4);
  float*  FT   = (float*)alloc(BCN * 4);
  float*  X3   = (float*)alloc(BCN * 4);
  __bf16* HB   = (__bf16*)alloc(BCN * 2);
  __bf16* HWB  = (__bf16*)alloc(BCN * 2);
  __bf16* MB   = (__bf16*)alloc(BCN * 2);
  __bf16* BCNK = (__bf16*)alloc(BNK * 2);
  __bf16* BSNK = (__bf16*)alloc(BNK * 2);
  __bf16* BCKN = (__bf16*)alloc(BNK * 2);
  __bf16* BSKN = (__bf16*)alloc(BNK * 2);
  float*  XC   = (float*)alloc(BCK * 4);
  float*  XS   = (float*)alloc(BCK * 4);
  float*  X0   = (float*)alloc((long)BB * CCH * 4);
  float*  F0   = (float*)alloc((long)BB * CCH * 4);
  __bf16* FCH  = (__bf16*)alloc(BCK * 2);
  __bf16* FSH  = (__bf16*)alloc(BCK * 2);
  __bf16* WWb  = (__bf16*)alloc((long)LLAY * CCH * CCH * 2);
  __bf16* G1b  = (__bf16*)alloc((long)LLAY * CCH * CCH * 2);
  __bf16* G2b  = (__bf16*)alloc((long)LLAY * CCH * CCH * 2);
  __bf16* F1T  = (__bf16*)alloc((long)CCH * FCF * 2);

  const int TB = 256;
  const long LW = (long)LLAY * CCH * CCH;

  // -------- init: weight conversions, bases, fc0 --------
  k_cvt<<<(unsigned)((LW + TB - 1) / TB), TB, 0, stream>>>(w_w, WWb, LW);
  k_cvt<<<(unsigned)((LW + TB - 1) / TB), TB, 0, stream>>>(g_w1, G1b, LW);
  k_cvt<<<(unsigned)((LW + TB - 1) / TB), TB, 0, stream>>>(g_w2, G2b, LW);
  k_cvt_fc1T<<<(CCH * FCF + TB - 1) / TB, TB, 0, stream>>>(fc1_w, F1T);
  k_bases<<<dim3(NNODE / 256, BB), 256, 0, stream>>>(nodes, mask, modes,
                                                     BCNK, BSNK, BCKN, BSKN);
  k_fc0<<<(unsigned)((BCN + TB - 1) / TB), TB, 0, stream>>>(x, fc0_w, fc0_b, H);

  const int SPLIT = 64;  // split-K for the skinny analysis GEMM
  const long sCN = (long)CCH * NNODE;
  const long sNK = (long)NNODE * KMODE;
  const long sCK = (long)CCH * KMODE;
  const long sKN = (long)KMODE * NNODE;

  for (int l = 0; l < LLAY; ++l) {
    // h -> bf16 (and node-weight scaled copy for analysis)
    k_cvt_h<<<(unsigned)((BCN + TB - 1) / TB), TB, 0, stream>>>(H, nw, HB, HWB, BCN);

    // analysis: xc = hw . bases_c ; xs = -(hw . bases_s)   (split-K + atomics)
    k_zero<<<256, TB, 0, stream>>>(XC, BCK);
    k_zero<<<256, TB, 0, stream>>>(XS, BCK);
    gemm_bf16<<<dim3(KMODE / GBN, CCH / GBM, BB * SPLIT), 128, 0, stream>>>(
        HWB, BCNK, XC, nullptr, CCH, KMODE, NNODE, sCN, sNK, sCK,
        1.0f, 2, 0, SPLIT);
    gemm_bf16<<<dim3(KMODE / GBN, CCH / GBM, BB * SPLIT), 128, 0, stream>>>(
        HWB, BSNK, XS, nullptr, CCH, KMODE, NNODE, sCN, sNK, sCK,
        -1.0f, 2, 0, SPLIT);
    k_x0<<<dim3(CCH, BB), 256, 0, stream>>>(H, nw, mask, X0);

    // spectral weight mix (tiny)
    k_specmix<<<dim3(KMODE, BB), CCH, 0, stream>>>(
        XC, XS, sp_wc + (long)l * CCH * CCH * KMODE,
        sp_ws + (long)l * CCH * CCH * KMODE, FCH, FSH);
    k_f0<<<BB, CCH, 0, stream>>>(X0, sp_w0 + (long)l * CCH * CCH, F0);

    // synthesis: D1 = 2*fc_h.bases_c^T - 2*fs_h.bases_s^T
    gemm_bf16<<<dim3(NNODE / GBN, CCH / GBM, BB), 128, 0, stream>>>(
        FCH, BCKN, D1, nullptr, CCH, NNODE, KMODE, sCK, sKN, sCN,
        2.0f, 0, 0, 1);
    gemm_bf16<<<dim3(NNODE / GBN, CCH / GBM, BB), 128, 0, stream>>>(
        FSH, BSKN, D1, nullptr, CCH, NNODE, KMODE, sCK, sKN, sCN,
        -2.0f, 1, 0, 1);
    // pointwise: D1 += w_w . h + w_b
    gemm_bf16<<<dim3(NNODE / GBN, CCH / GBM, BB), 128, 0, stream>>>(
        WWb + (long)l * CCH * CCH, HB, D1, w_b + l * CCH,
        CCH, NNODE, CCH, 0, sCN, sCN, 1.0f, 1, 0, 1);

    if (l != LLAY - 1) {
      // graph branch: m = gelu(g1.h+b1); f = g2.m+b2; scatter Gaussian messages
      gemm_bf16<<<dim3(NNODE / GBN, CCH / GBM, BB), 128, 0, stream>>>(
          G1b + (long)l * CCH * CCH, HB, MBUF, g_b1 + l * CCH,
          CCH, NNODE, CCH, 0, sCN, sCN, 1.0f, 0, 1, 1);
      k_cvt<<<(unsigned)((BCN + TB - 1) / TB), TB, 0, stream>>>(MBUF, MB, BCN);
      gemm_bf16<<<dim3(NNODE / GBN, CCH / GBM, BB), 128, 0, stream>>>(
          G2b + (long)l * CCH * CCH, MB, FBUF, g_b2 + l * CCH,
          CCH, NNODE, CCH, 0, sCN, sCN, 1.0f, 0, 0, 1);
      k_transpose<<<dim3(NNODE / 32, CCH / 32, BB), dim3(32, 8), 0, stream>>>(FBUF, FT);
      k_zero<<<4096, TB, 0, stream>>>(X3, BCN);
      k_edges<<<dim3(EEDGE, BB), CCH, 0, stream>>>(
          edges, nodes, FT, isw + l * 4, isb + l * 2, gC + l, X3);
      k_combine<<<(unsigned)((BCN + TB - 1) / TB), TB, 0, stream>>>(
          D1, F0, mask, X3, H, 1);
    } else {
      k_combine<<<(unsigned)((BCN + TB - 1) / TB), TB, 0, stream>>>(
          D1, F0, mask, nullptr, H, 0);
    }
  }

  // final: h1 = gelu(fc1^T . h + fc1_b); out = h1^T . fc2_w + fc2_b
  k_cvt<<<(unsigned)((BCN + TB - 1) / TB), TB, 0, stream>>>(H, HB, BCN);
  gemm_bf16<<<dim3(NNODE / GBN, FCF / GBM, BB), 128, 0, stream>>>(
      F1T, HB, MBUF, fc1_b, FCF, NNODE, CCH, 0, sCN, (long)FCF * NNODE,
      1.0f, 0, 1, 1);
  k_out<<<dim3(NNODE / 256, BB), 256, 0, stream>>>(MBUF, fc2_w, fc2_b, out);
}
